// SelfAttentionBlock_5772436046513
// MI455X (gfx1250) — matrix operand order repaired and checked by reading, not compiled, not run on hardware
//
#include <hip/hip_runtime.h>

// MI455X / gfx1250, wave32. Flash-attention with v_wmma_f32_16x16x32_f16,
// double-buffered LDS staging via GLOBAL_LOAD_ASYNC_TO_LDS_B128 (ASYNCcnt).
//
// Shapes: B=4, C=256, CK=128, H=W=64 -> N=4096.
// ~56 GFLOP total vs ~80 MB HBM traffic -> compute bound; all MACs go
// through WMMA f16 (f32 accum). K+V f16 per batch = 3 MB -> whole working
// set lives in the 192 MB L2; K/V 32-key blocks are staged in LDS shared by
// 4 waves/workgroup and prefetched one block ahead asynchronously.

typedef __attribute__((ext_vector_type(16))) _Float16 v16h;
typedef __attribute__((ext_vector_type(8)))  _Float16 v8h;
typedef __attribute__((ext_vector_type(8)))  float    v8f;
typedef __attribute__((ext_vector_type(4)))  int      v4i;

#define B_   4
#define C_   256
#define CK_  128
#define N_   4096
#define KB_  32            // keys per flash block
#define QT_  16            // queries per wave
#define WV_  4             // waves per workgroup in attention kernel
#define NB_  (N_ / KB_)    // key blocks

#if defined(__AMDGCN__) && __has_builtin(__builtin_amdgcn_global_load_async_to_lds_b128)
#define ASYNC_LDS 1
#else
#define ASYNC_LDS 0
#endif

// ---------------------------------------------------------------------------
// Projection kernel: q = Wq x + bq, k = Wk x + bk, v = Wv x + bv.
// One wave handles 16 pixels; x B-fragments (full C=256 reduction) stay
// resident in VGPRs while we sweep 32 output-row tiles (8 WMMAs each).
//   Qh,Kh layout: [b][n][128]  (f16)  -> contiguous frag loads in attention
//   Vh layout:    [b][c][4096] (f16)  -> contiguous V B-frag loads
// ---------------------------------------------------------------------------
__global__ __launch_bounds__(32) void proj_kernel(
    const float* __restrict__ x,
    const float* __restrict__ Wq, const float* __restrict__ bq,
    const float* __restrict__ Wk, const float* __restrict__ bk,
    const float* __restrict__ Wv, const float* __restrict__ bv,
    _Float16* __restrict__ Qh, _Float16* __restrict__ Kh,
    _Float16* __restrict__ Vh)
{
    const int wg   = blockIdx.x;            // 0 .. B*N/16-1
    const int b    = wg / (N_ / 16);
    const int n0   = (wg % (N_ / 16)) * 16;
    const int lane = threadIdx.x & 31;
    const int lh   = lane & 15;
    const int hi   = lane >> 4;             // 0 or 1

    const float* xb = x + (size_t)b * C_ * N_;
    const int n = n0 + lh;

    // B-fragments of x: lane col = pixel, halves h -> in_ch = c*32+hi*8+(h&7)+(h>>3)*16.
    v16h xf[8];
#pragma unroll
    for (int c = 0; c < 8; ++c) {
        _Float16* ap = reinterpret_cast<_Float16*>(&xf[c]);
#pragma unroll
        for (int h = 0; h < 16; ++h) {
            int ic = c * 32 + hi * 8 + (h & 7) + (h >> 3) * 16;
            ap[h] = (_Float16)xb[(size_t)ic * N_ + n];
        }
    }

    for (int t = 0; t < 32; ++t) {
        const float* Wm; const float* bias; int m0;
        if (t < 8)       { Wm = Wq; bias = bq; m0 = t * 16; }
        else if (t < 16) { Wm = Wk; bias = bk; m0 = (t - 8) * 16; }
        else             { Wm = Wv; bias = bv; m0 = (t - 16) * 16; }

        v8f acc = {};
#pragma unroll
        for (int c = 0; c < 8; ++c) {
            // A-fragment of W: lane row M = lh -> out_ch m0+lh.
            v16h af;
            _Float16* ap = reinterpret_cast<_Float16*>(&af);
            const float* wr = Wm + (size_t)(m0 + lh) * C_ + c * 32 + hi * 8;
#pragma unroll
            for (int h = 0; h < 8; ++h) ap[h]     = (_Float16)wr[h];
#pragma unroll
            for (int h = 0; h < 8; ++h) ap[8 + h] = (_Float16)wr[16 + h];
            acc = __builtin_amdgcn_wmma_f32_16x16x32_f16(
                false, af, false, xf[c], (short)0, acc, false, false);
        }

        const int mrow = m0 + hi * 8;
        if (t < 16) {
            _Float16* dst = (t < 8 ? Qh : Kh) +
                ((size_t)b * N_ + n) * CK_ + mrow;
            v8h tmp;
#pragma unroll
            for (int r = 0; r < 8; ++r)
                tmp[r] = (_Float16)(acc[r] + bias[mrow + r]);
            *reinterpret_cast<v8h*>(dst) = tmp;      // 16B store
        } else {
            _Float16* dstb = Vh + (size_t)b * C_ * N_;
#pragma unroll
            for (int r = 0; r < 8; ++r)
                dstb[(size_t)(mrow + r) * N_ + n] =
                    (_Float16)(acc[r] + bias[mrow + r]);
        }
    }
}

// ---------------------------------------------------------------------------
// Flash-attention kernel. 4 waves/workgroup, each wave owns 16 queries and
// the full d_v=256 accumulator (16 x v8f). K/V 32-key blocks are staged in
// double-buffered LDS, prefetched one block ahead via async-to-LDS.
// Online softmax, unscaled (per reference).
// ---------------------------------------------------------------------------
__global__ __launch_bounds__(WV_ * 32) void attn_kernel(
    const float* __restrict__ x, const float* __restrict__ gamma,
    const _Float16* __restrict__ Qh, const _Float16* __restrict__ Kh,
    const _Float16* __restrict__ Vh, float* __restrict__ out)
{
    __shared__ _Float16 lK[2][KB_][CK_];       // 2 x 8 KB : keys x d
    __shared__ _Float16 lV[2][C_][KB_];        // 2 x 16 KB: channels x keys
    __shared__ _Float16 lP[WV_][QT_][KB_];     // 4 KB : per-wave P bounce

    const int blocksPerBatch = N_ / (QT_ * WV_);
    const int b    = blockIdx.x / blocksPerBatch;
    const int q0wg = (blockIdx.x % blocksPerBatch) * (QT_ * WV_);
    const int tid  = threadIdx.x;
    const int wid  = tid >> 5;
    const int lane = tid & 31;
    const int lh   = lane & 15;
    const int hi   = lane >> 4;
    const int q0   = q0wg + wid * QT_;

    const _Float16* Qb = Qh + (size_t)b * N_ * CK_;
    const _Float16* Kb = Kh + (size_t)b * N_ * CK_;
    const _Float16* Vb = Vh + (size_t)b * C_ * N_;

    // Stage one 32-key K/V block into LDS buffer `buf`.
    // Per thread: 4 K b128s + 8 V b128s = 12 transfers.
    auto stage = [&](int m0, int buf) {
#if ASYNC_LDS
        typedef __attribute__((address_space(1))) v4i GV4;   // global
        typedef __attribute__((address_space(3))) v4i LV4;   // LDS
        const uint4* ks = reinterpret_cast<const uint4*>(Kb + (size_t)m0 * CK_);
        uint4* kd = reinterpret_cast<uint4*>(&lK[buf][0][0]);
        uint4* vd = reinterpret_cast<uint4*>(&lV[buf][0][0]);
#pragma unroll
        for (int i = 0; i < 4; ++i) {
            int idx = i * 128 + tid;
            __builtin_amdgcn_global_load_async_to_lds_b128(
                (GV4*)(ks + idx), (LV4*)(kd + idx), 0, 0);
        }
#pragma unroll
        for (int i = 0; i < 8; ++i) {
            int idx  = i * 128 + tid;
            int ch   = idx >> 2;
            int part = idx & 3;
            __builtin_amdgcn_global_load_async_to_lds_b128(
                (GV4*)(Vb + (size_t)ch * N_ + m0 + part * 8),
                (LV4*)(vd + idx), 0, 0);
        }
#else
        uint4* kd = reinterpret_cast<uint4*>(&lK[buf][0][0]);
        const uint4* ks = reinterpret_cast<const uint4*>(Kb + (size_t)m0 * CK_);
#pragma unroll
        for (int i = 0; i < 4; ++i) kd[i * 128 + tid] = ks[i * 128 + tid];
        uint4* vd = reinterpret_cast<uint4*>(&lV[buf][0][0]);
#pragma unroll
        for (int i = 0; i < 8; ++i) {
            int idx  = i * 128 + tid;
            int ch   = idx >> 2;
            int part = idx & 3;
            vd[idx] = *reinterpret_cast<const uint4*>(
                Vb + (size_t)ch * N_ + m0 + part * 8);
        }
#endif
    };

    // Resident Q A-fragments: 4 chunks of 32 d-dims.
    v16h qf[4];
#pragma unroll
    for (int c = 0; c < 4; ++c) {
        const _Float16* src = Qb + (size_t)(q0 + lh) * CK_ + c * 32 + hi * 16;
        _Float16* ap = reinterpret_cast<_Float16*>(&qf[c]);
        *reinterpret_cast<v8h*>(ap)     = *reinterpret_cast<const v8h*>(src);
        *reinterpret_cast<v8h*>(ap + 8) = *reinterpret_cast<const v8h*>(src + 8);
    }

    v8f o[16];
    v8f zero = {};
#pragma unroll
    for (int i = 0; i < 16; ++i) o[i] = zero;
    float mrow[8], lrow[8];
#pragma unroll
    for (int r = 0; r < 8; ++r) { mrow[r] = -3.0e38f; lrow[r] = 0.0f; }

    stage(0, 0);                               // prologue prefetch

    for (int blk = 0; blk < NB_; ++blk) {
        const int cur = blk & 1;
        __syncthreads();            // all waves done reading buffer cur^1
        if (blk + 1 < NB_) {
            stage((blk + 1) * KB_, cur ^ 1);   // prefetch next block
#if ASYNC_LDS
            // 12 outstanding = next block's transfers; async loads complete
            // in order, so this guarantees the current block has landed.
            asm volatile("s_wait_asynccnt 0xc" ::: "memory");
#endif
        } else {
#if ASYNC_LDS
            asm volatile("s_wait_asynccnt 0x0" ::: "memory");
#endif
        }
        __syncthreads();            // current block visible to all waves

        // --- S = Q K^T : two 16-key subtiles, 4 d-chunks each ---
        v8f s0 = zero, s1 = zero;
#pragma unroll
        for (int c = 0; c < 4; ++c) {
            v16h kf0, kf1;
            const _Float16* p0 = &lK[cur][lh][c * 32 + hi * 16];
            reinterpret_cast<uint4*>(&kf0)[0] = reinterpret_cast<const uint4*>(p0)[0];
            reinterpret_cast<uint4*>(&kf0)[1] = reinterpret_cast<const uint4*>(p0)[1];
            const _Float16* p1 = &lK[cur][16 + lh][c * 32 + hi * 16];
            reinterpret_cast<uint4*>(&kf1)[0] = reinterpret_cast<const uint4*>(p1)[0];
            reinterpret_cast<uint4*>(&kf1)[1] = reinterpret_cast<const uint4*>(p1)[1];
            s0 = __builtin_amdgcn_wmma_f32_16x16x32_f16(
                false, qf[c], false, kf0, (short)0, s0, false, false);
            s1 = __builtin_amdgcn_wmma_f32_16x16x32_f16(
                false, qf[c], false, kf1, (short)0, s1, false, false);
        }

        // --- online softmax (each row lives in one 16-lane group) ---
#pragma unroll
        for (int r = 0; r < 8; ++r) {
            float mx = fmaxf(s0[r], s1[r]);
#pragma unroll
            for (int off = 8; off >= 1; off >>= 1)
                mx = fmaxf(mx, __shfl_xor(mx, off, 32));
            float mnew = fmaxf(mrow[r], mx);
            float a  = __expf(mrow[r] - mnew);
            float e0 = __expf(s0[r] - mnew);
            float e1 = __expf(s1[r] - mnew);
            s0[r] = e0; s1[r] = e1;
            float rs = e0 + e1;
#pragma unroll
            for (int off = 8; off >= 1; off >>= 1)
                rs += __shfl_xor(rs, off, 32);
            lrow[r] = lrow[r] * a + rs;
            mrow[r] = mnew;
#pragma unroll
            for (int ct = 0; ct < 16; ++ct) o[ct][r] *= a;
        }

        // --- P: D layout -> LDS [row][key] -> A fragment (wave-private) ---
#pragma unroll
        for (int r = 0; r < 8; ++r) {
            lP[wid][r + hi * 8][lh]      = (_Float16)s0[r];
            lP[wid][r + hi * 8][16 + lh] = (_Float16)s1[r];
        }
        v16h pf;
        {
            const _Float16* src = &lP[wid][lh][hi * 16];
            _Float16* ap = reinterpret_cast<_Float16*>(&pf);
            *reinterpret_cast<v8h*>(ap)     = *reinterpret_cast<const v8h*>(src);
            *reinterpret_cast<v8h*>(ap + 8) = *reinterpret_cast<const v8h*>(src + 8);
        }

        // --- O += P * V : 16 channel tiles ---
#pragma unroll
        for (int ct = 0; ct < 16; ++ct) {
            v16h vf;
            const _Float16* vp = &lV[cur][ct * 16 + lh][hi * 16];
            reinterpret_cast<uint4*>(&vf)[0] = reinterpret_cast<const uint4*>(vp)[0];
            reinterpret_cast<uint4*>(&vf)[1] = reinterpret_cast<const uint4*>(vp)[1];
            o[ct] = __builtin_amdgcn_wmma_f32_16x16x32_f16(
                false, pf, false, vf, (short)0, o[ct], false, false);
        }
    }

    // --- epilogue: out = gamma * (O / l) + x ---
    const float g = gamma[0];
    const float* xb = x   + (size_t)b * C_ * N_;
    float*       ob = out + (size_t)b * C_ * N_;
    float inv[8];
#pragma unroll
    for (int r = 0; r < 8; ++r) inv[r] = 1.0f / lrow[r];
#pragma unroll
    for (int ct = 0; ct < 16; ++ct) {
        int ch = ct * 16 + lh;                   // D col = channel
        float*       op = ob + (size_t)ch * N_ + q0 + hi * 8;
        const float* xp = xb + (size_t)ch * N_ + q0 + hi * 8;
#pragma unroll
        for (int r = 0; r < 8; ++r)              // D row = query q0+hi*8+r
            op[r] = g * (o[ct][r] * inv[r]) + xp[r];
    }
}

// ---------------------------------------------------------------------------
extern "C" void kernel_launch(void* const* d_in, const int* in_sizes, int n_in,
                              void* d_out, int out_size, void* d_ws, size_t ws_size,
                              hipStream_t stream) {
    (void)in_sizes; (void)n_in; (void)out_size; (void)ws_size;
    const float* x     = (const float*)d_in[0];
    const float* Wq    = (const float*)d_in[1];
    const float* bq    = (const float*)d_in[2];
    const float* Wk    = (const float*)d_in[3];
    const float* bk    = (const float*)d_in[4];
    const float* Wv    = (const float*)d_in[5];
    const float* bv    = (const float*)d_in[6];
    const float* gamma = (const float*)d_in[7];
    float* out = (float*)d_out;

    _Float16* Qh = (_Float16*)d_ws;                       // 4 MB
    _Float16* Kh = Qh + (size_t)B_ * N_ * CK_;            // 4 MB
    _Float16* Vh = Kh + (size_t)B_ * N_ * CK_;            // 8 MB

    proj_kernel<<<dim3(B_ * N_ / 16), dim3(32), 0, stream>>>(
        x, Wq, bq, Wk, bk, Wv, bv, Qh, Kh, Vh);
    attn_kernel<<<dim3(B_ * (N_ / (QT_ * WV_))), dim3(WV_ * 32), 0, stream>>>(
        x, gamma, Qh, Kh, Vh, out);
}
